// SelfAggregationModule_21930103013655
// MI455X (gfx1250) — compile-verified, matrix operand order checked by reading, not run on hardware
//
#include <hip/hip_runtime.h>
#include <hip/hip_bf16.h>

#define NTOK 147456   // 96*96*16
#define CCH  128
#define TCH  384
#define HEADS 8
#define FFN  512

typedef __attribute__((ext_vector_type(16))) __bf16 v16bf;
typedef __attribute__((ext_vector_type(8)))  float  v8f;
typedef __attribute__((ext_vector_type(8)))  unsigned short us8;

union BF16Frag { v16bf v; us8 h[2]; unsigned short u[16]; };

__device__ __forceinline__ unsigned short f2bf(float f) {
    unsigned int u = __float_as_uint(f);
    u = (u + 0x7FFFu + ((u >> 16) & 1u)) >> 16;
    return (unsigned short)u;
}
__device__ __forceinline__ float bf2f(unsigned short u) {
    return __uint_as_float(((unsigned int)u) << 16);
}

// Inverse of the wave32 16-bit A/B fragment lane layout:
// lane L = hi*16 + idx16, element j -> k_local = (j<8 ? j+hi*8 : 16+(j-8)+hi*8)
__device__ __forceinline__ int frag_idx(int s, int kl, int t) {
    int base = (kl < 16) ? 0 : 8;
    int m    = (kl < 16) ? kl : (kl - 16);
    int hi   = (m >= 8) ? 1 : 0;
    int j    = base + (m - hi * 8);
    return ((s * 32 + hi * 16 + t) * 16 + j);
}

__device__ __forceinline__ v16bf load16(const unsigned short* p) {
    BF16Frag f;
    f.h[0] = *(const us8*)(p);
    f.h[1] = *(const us8*)(p + 8);
    return f.v;
}

__device__ __forceinline__ v8f wmma_bf16(v16bf a, v16bf b, v8f c) {
    return __builtin_amdgcn_wmma_f32_16x16x32_bf16(false, a, false, b,
                                                   (short)0, c, false, false);
}

// ---------------- per-channel mean / rstd over N (instance norm stats) -----
__global__ void stats_kernel(const float* __restrict__ src,
                             float* __restrict__ m_out, float* __restrict__ r_out) {
    int c = blockIdx.x;
    const float* p = src + (size_t)c * NTOK;
    float s = 0.f, s2 = 0.f;
    for (int i = threadIdx.x; i < NTOK; i += 256) { float v = p[i]; s += v; s2 += v * v; }
    __shared__ float sh[512];
    sh[threadIdx.x] = s; sh[256 + threadIdx.x] = s2;
    __syncthreads();
    for (int o = 128; o > 0; o >>= 1) {
        if ((int)threadIdx.x < o) {
            sh[threadIdx.x] += sh[threadIdx.x + o];
            sh[256 + threadIdx.x] += sh[256 + threadIdx.x + o];
        }
        __syncthreads();
    }
    if (threadIdx.x == 0) {
        float m = sh[0] / (float)NTOK;
        float v = sh[256] / (float)NTOK - m * m;
        m_out[c] = m; r_out[c] = rsqrtf(v + 1e-5f);
    }
}

// ---------------- weight pre-swizzle (f32 -> bf16 fragment order) ----------
__global__ void swizzle_A(const float* __restrict__ W, unsigned short* __restrict__ dst,
                          int ktiles, int ldk) {
    int bx = blockIdx.x, ot = bx / ktiles, s = bx % ktiles;
    int e = threadIdx.x, L = e >> 4, j = e & 15, hi = L >> 4;
    int kl = (j < 8) ? (j + hi * 8) : (16 + (j - 8) + hi * 8);
    dst[(size_t)bx * 512 + e] = f2bf(W[(ot * 16 + (L & 15)) * ldk + s * 32 + kl]);
}
__global__ void swizzle_B(const float* __restrict__ W, unsigned short* __restrict__ dst,
                          int ktiles, int ncols) {
    int bx = blockIdx.x, nt = bx / ktiles, s = bx % ktiles;
    int e = threadIdx.x, L = e >> 4, j = e & 15, hi = L >> 4;
    int kl = (j < 8) ? (j + hi * 8) : (16 + (j - 8) + hi * 8);
    dst[(size_t)bx * 512 + e] = f2bf(W[(s * 32 + kl) * ncols + nt * 16 + (L & 15)]);
}

// ---------------- fused instnorm + qkv 1x1-conv GEMM (384x128 @ 128xN) -----
__global__ void qkv_gemm_kernel(const float* __restrict__ x,
                                const float* __restrict__ m1, const float* __restrict__ r1,
                                const unsigned short* __restrict__ Wsw,
                                unsigned short* __restrict__ qkv0) {
    __shared__ unsigned short Bl[2048];
    int tokbase = blockIdx.x * 16;
    int tid = threadIdx.x;
    {   // stage normalized x tile (128ch x 16tok) pre-swizzled as B operand
        int k = tid >> 1, t0 = (tid & 1) * 8;
        const float* xp = x + (size_t)k * NTOK + tokbase + t0;
        float m = m1[k], r = r1[k];
        float4 a = ((const float4*)xp)[0];
        float4 b = ((const float4*)(xp + 4))[0];
        float vv[8] = {a.x, a.y, a.z, a.w, b.x, b.y, b.z, b.w};
        int s = k >> 5, kl = k & 31;
#pragma unroll
        for (int e = 0; e < 8; e++)
            Bl[frag_idx(s, kl, t0 + e)] = f2bf((vv[e] - m) * r);
    }
    __syncthreads();
    int w = tid >> 5, lane = tid & 31, hi = lane >> 4, li = lane & 15;
#pragma unroll
    for (int oo = 0; oo < 3; oo++) {
        int ot = w + oo * 8;
        v8f acc = {0.f, 0.f, 0.f, 0.f, 0.f, 0.f, 0.f, 0.f};
#pragma unroll
        for (int s = 0; s < 4; s++) {
            v16bf a = load16(Wsw + ((size_t)(ot * 4 + s) * 32 + lane) * 16);
            v16bf b = load16(Bl + (s * 32 + lane) * 16);
            acc = wmma_bf16(a, b, acc);
        }
#pragma unroll
        for (int rr = 0; rr < 8; rr++) {
            int o = ot * 16 + rr + hi * 8;
            qkv0[(size_t)o * NTOK + tokbase + li] = f2bf(acc[rr]);
        }
    }
}

// ---------------- depthwise 3x3x3 conv + q/k sum-of-squares ----------------
__global__ void dwconv_kernel(const unsigned short* __restrict__ qkv0,
                              const float* __restrict__ Wdw,
                              unsigned short* __restrict__ qkv1,
                              float* __restrict__ norms) {
    int ch = blockIdx.x / 576;
    int nl = (blockIdx.x % 576) * 256 + threadIdx.x;
    int zz = nl & 15, ww = (nl >> 4) % 96, hh = nl / 1536;
    const unsigned short* src = qkv0 + (size_t)ch * NTOK;
    const float* wp = Wdw + ch * 27;
    float acc = 0.f;
#pragma unroll
    for (int dh = -1; dh <= 1; dh++)
#pragma unroll
        for (int dw = -1; dw <= 1; dw++)
#pragma unroll
            for (int dz = -1; dz <= 1; dz++) {
                int h2 = hh + dh, w2 = ww + dw, z2 = zz + dz;
                if (h2 >= 0 && h2 < 96 && w2 >= 0 && w2 < 96 && z2 >= 0 && z2 < 16)
                    acc += wp[((dh + 1) * 3 + (dw + 1)) * 3 + (dz + 1)] *
                           bf2f(src[(h2 * 96 + w2) * 16 + z2]);
            }
    qkv1[(size_t)ch * NTOK + nl] = f2bf(acc);
    __shared__ float red[256];
    red[threadIdx.x] = (ch < 256) ? acc * acc : 0.f;
    __syncthreads();
    for (int o = 128; o > 0; o >>= 1) {
        if ((int)threadIdx.x < o) red[threadIdx.x] += red[threadIdx.x + o];
        __syncthreads();
    }
    if (threadIdx.x == 0 && ch < 256) atomicAdd(&norms[ch], red[0]);
}

// ---------------- raw Gram S = q . k^T per head (WMMA over N) --------------
__global__ void attn_accum_kernel(const unsigned short* __restrict__ qkv1,
                                  float* __restrict__ S) {
    int h = blockIdx.y;
    int w = threadIdx.x >> 5, lane = threadIdx.x & 31, hi = lane >> 4, li = lane & 15;
    const unsigned short* qrow = qkv1 + (size_t)(h * 16 + li) * NTOK;
    const unsigned short* krow = qkv1 + (size_t)(128 + h * 16 + li) * NTOK;
    v8f acc = {0.f, 0.f, 0.f, 0.f, 0.f, 0.f, 0.f, 0.f};
    int base = (blockIdx.x * 8 + w) * 2048;
    for (int it = 0; it < 64; it++) {
        int nb = base + it * 32;
        __builtin_prefetch(qrow + nb + 256, 0, 0);
        __builtin_prefetch(krow + nb + 256, 0, 0);
        BF16Frag fa, fb;
        fa.h[0] = *(const us8*)(qrow + nb + hi * 8);
        fa.h[1] = *(const us8*)(qrow + nb + 16 + hi * 8);
        fb.h[0] = *(const us8*)(krow + nb + hi * 8);
        fb.h[1] = *(const us8*)(krow + nb + 16 + hi * 8);
        acc = wmma_bf16(fa.v, fb.v, acc);
    }
#pragma unroll
    for (int r = 0; r < 8; r++)
        atomicAdd(&S[h * 256 + (r + hi * 8) * 16 + li], acc[r]);
}

// ---------------- softmax with l2-norm + temperature folded ----------------
__global__ void softmax_kernel(const float* __restrict__ S, const float* __restrict__ norms,
                               const float* __restrict__ temp, float* __restrict__ attn) {
    int t = threadIdx.x;
    int h = t >> 4, i = t & 15;
    float nq = fmaxf(sqrtf(norms[h * 16 + i]), 1e-12f);
    float T = temp[h];
    float lg[16]; float mx = -1e30f;
#pragma unroll
    for (int j = 0; j < 16; j++) {
        float nk = fmaxf(sqrtf(norms[128 + h * 16 + j]), 1e-12f);
        float v = S[h * 256 + i * 16 + j] / (nq * nk) * T;
        lg[j] = v; mx = fmaxf(mx, v);
    }
    float sm = 0.f;
#pragma unroll
    for (int j = 0; j < 16; j++) { lg[j] = __expf(lg[j] - mx); sm += lg[j]; }
#pragma unroll
    for (int j = 0; j < 16; j++) attn[h * 256 + i * 16 + j] = lg[j] / sm;
}

// ---------------- fused attn @ v  ->  proj GEMM  + residual ----------------
__global__ void av_proj_kernel(const unsigned short* __restrict__ qkv1,
                               const float* __restrict__ attn_,
                               const unsigned short* __restrict__ Wproj_sw,
                               const float* __restrict__ x, float* __restrict__ y) {
    __shared__ unsigned short o1[2048];
    int tokbase = blockIdx.x * 16;
    int w = threadIdx.x >> 5, lane = threadIdx.x & 31, hi = lane >> 4, li = lane & 15;
    {   // step 1: per-head o1 = attn[h] (16x16, K padded to 32) @ v tile
        int h = w;
        BF16Frag fa, fb;
#pragma unroll
        for (int j = 0; j < 16; j++) {
            int kl = (j < 8) ? (j + hi * 8) : (16 + (j - 8) + hi * 8);
            fa.u[j] = (kl < 16) ? f2bf(attn_[h * 256 + li * 16 + kl]) : (unsigned short)0;
            fb.u[j] = (kl < 16)
                ? qkv1[(size_t)(256 + h * 16 + kl) * NTOK + tokbase + li]
                : (unsigned short)0;
        }
        v8f acc = {0.f, 0.f, 0.f, 0.f, 0.f, 0.f, 0.f, 0.f};
        acc = wmma_bf16(fa.v, fb.v, acc);
#pragma unroll
        for (int r = 0; r < 8; r++) {
            int cp = h * 16 + r + hi * 8;
            o1[frag_idx(cp >> 5, cp & 31, li)] = f2bf(acc[r]);
        }
    }
    __syncthreads();
    {   // step 2: out-channel tile ot = wave: Wproj (A) x o1 (B from LDS)
        int ot = w;
        v8f acc = {0.f, 0.f, 0.f, 0.f, 0.f, 0.f, 0.f, 0.f};
#pragma unroll
        for (int s = 0; s < 4; s++) {
            v16bf a = load16(Wproj_sw + ((size_t)(ot * 4 + s) * 32 + lane) * 16);
            v16bf b = load16(o1 + (s * 32 + lane) * 16);
            acc = wmma_bf16(a, b, acc);
        }
#pragma unroll
        for (int r = 0; r < 8; r++) {
            int c = ot * 16 + r + hi * 8;
            size_t idx = (size_t)c * NTOK + tokbase + li;
            y[idx] = x[idx] + acc[r];
        }
    }
}

// ---------------- fused FFN: norm -> GEMM1 -> relu -> GEMM2 -> residual ----
__global__ void ffn_kernel(const float* __restrict__ y,
                           const float* __restrict__ m2, const float* __restrict__ r2,
                           const unsigned short* __restrict__ W1sw, const float* __restrict__ b1,
                           const unsigned short* __restrict__ W2sw, const float* __restrict__ b2,
                           const float* __restrict__ x, float* __restrict__ out) {
    __shared__ unsigned short tokA[2048];
    __shared__ unsigned short hA[8192];
    int tokbase = blockIdx.x * 16;
    int tid = threadIdx.x;
    {   // stage normalized token tile (16tok x 128ch) as A operand
        int k = tid >> 1, t0 = (tid & 1) * 8;
        const float* yp = y + (size_t)k * NTOK + tokbase + t0;
        float m = m2[k], r = r2[k];
        float4 a = ((const float4*)yp)[0];
        float4 b = ((const float4*)(yp + 4))[0];
        float vv[8] = {a.x, a.y, a.z, a.w, b.x, b.y, b.z, b.w};
        int s = k >> 5, kl = k & 31;
#pragma unroll
        for (int e = 0; e < 8; e++)
            tokA[frag_idx(s, kl, t0 + e)] = f2bf((vv[e] - m) * r);
    }
    __syncthreads();
    int w = tid >> 5, lane = tid & 31, hi = lane >> 4, li = lane & 15;
#pragma unroll
    for (int q = 0; q < 4; q++) {   // 32 hidden tiles / 8 waves
        int ht = w + q * 8;
        v8f acc = {0.f, 0.f, 0.f, 0.f, 0.f, 0.f, 0.f, 0.f};
#pragma unroll
        for (int s = 0; s < 4; s++) {
            v16bf a = load16(tokA + (s * 32 + lane) * 16);
            v16bf b = load16(W1sw + ((size_t)(ht * 4 + s) * 32 + lane) * 16);
            acc = wmma_bf16(a, b, acc);
        }
#pragma unroll
        for (int r = 0; r < 8; r++) {
            int t = r + hi * 8, f = ht * 16 + li;
            float hv = fmaxf(acc[r] + b1[f], 0.f);
            hA[frag_idx(f >> 5, f & 31, t)] = f2bf(hv);
        }
    }
    __syncthreads();
    {   // second GEMM: hidden (K=512) x W2 -> out tile; add bias + x residual
        int ot = w;
        v8f acc = {0.f, 0.f, 0.f, 0.f, 0.f, 0.f, 0.f, 0.f};
#pragma unroll
        for (int s = 0; s < 16; s++) {
            v16bf a = load16(hA + (s * 32 + lane) * 16);
            v16bf b = load16(W2sw + ((size_t)(ot * 16 + s) * 32 + lane) * 16);
            acc = wmma_bf16(a, b, acc);
        }
#pragma unroll
        for (int r = 0; r < 8; r++) {
            int t = r + hi * 8, c = ot * 16 + li;
            size_t idx = (size_t)c * NTOK + tokbase + t;
            out[idx] = x[idx] + b2[c] + acc[r];
        }
    }
}

extern "C" void kernel_launch(void* const* d_in, const int* in_sizes, int n_in,
                              void* d_out, int out_size, void* d_ws, size_t ws_size,
                              hipStream_t stream) {
    (void)in_sizes; (void)n_in; (void)out_size; (void)ws_size;
    const float* x     = (const float*)d_in[0];
    const float* Wqkv  = (const float*)d_in[1];
    const float* Wdw   = (const float*)d_in[2];
    const float* temp  = (const float*)d_in[3];
    const float* Wproj = (const float*)d_in[4];
    const float* W1    = (const float*)d_in[5];
    const float* b1    = (const float*)d_in[6];
    const float* W2    = (const float*)d_in[7];
    const float* b2    = (const float*)d_in[8];

    char* ws = (char*)d_ws;
    float* m1    = (float*)(ws + 0);
    float* r1    = (float*)(ws + 512);
    float* m2    = (float*)(ws + 1024);
    float* r2    = (float*)(ws + 1536);
    float* norms = (float*)(ws + 2048);     // 256 f32 sum-of-squares (q,k rows)
    float* S     = (float*)(ws + 3072);     // 8*16*16 raw gram
    float* attn  = (float*)(ws + 11264);    // 8*16*16 softmaxed
    unsigned short* Wqkv_sw  = (unsigned short*)(ws + 20480);    // 96 KB
    unsigned short* Wproj_sw = (unsigned short*)(ws + 118784);   // 32 KB
    unsigned short* W1_sw    = (unsigned short*)(ws + 151552);   // 128 KB
    unsigned short* W2_sw    = (unsigned short*)(ws + 282624);   // 128 KB
    unsigned short* qkv0     = (unsigned short*)(ws + 413696);   // 113 MB bf16
    unsigned short* qkv1     = (unsigned short*)(ws + 113659904);// 113 MB bf16
    float* y = (float*)(ws + 413696);                            // alias qkv0 (free after dwconv)

    hipMemsetAsync(ws + 2048, 0, 9216, stream);  // zero norms + S accumulators

    swizzle_A<<<96, 512, 0, stream>>>(Wqkv, Wqkv_sw, 4, 128);
    swizzle_A<<<32, 512, 0, stream>>>(Wproj, Wproj_sw, 4, 128);
    swizzle_B<<<128, 512, 0, stream>>>(W1, W1_sw, 4, 512);
    swizzle_B<<<128, 512, 0, stream>>>(W2, W2_sw, 16, 128);

    stats_kernel<<<128, 256, 0, stream>>>(x, m1, r1);
    qkv_gemm_kernel<<<NTOK / 16, 256, 0, stream>>>(x, m1, r1, Wqkv_sw, qkv0);
    dwconv_kernel<<<TCH * 576, 256, 0, stream>>>(qkv0, Wdw, qkv1, norms);
    attn_accum_kernel<<<dim3(9, 8), 256, 0, stream>>>(qkv1, S);
    softmax_kernel<<<1, 128, 0, stream>>>(S, norms, temp, attn);
    av_proj_kernel<<<NTOK / 16, 256, 0, stream>>>(qkv1, attn, Wproj_sw, x, y);
    stats_kernel<<<128, 256, 0, stream>>>(y, m2, r2);
    ffn_kernel<<<NTOK / 16, 256, 0, stream>>>(y, m2, r2, W1_sw, b1, W2_sw, b2, x,
                                              (float*)d_out);
}
